// SEF_MGN_2_20023137534010
// MI455X (gfx1250) — matrix operand order repaired and checked by reading, not compiled, not run on hardware
//
#include <hip/hip_runtime.h>

#define N_NODES 10000
#define N_EDGES 80000
#define H 32

typedef __attribute__((ext_vector_type(16))) _Float16 v16h;
typedef __attribute__((ext_vector_type(8)))  float    v8f;

__device__ __forceinline__ float wave_reduce_sum(float v) {
    // wave32 butterfly: every lane ends with the full sum
    #pragma unroll
    for (int off = 16; off > 0; off >>= 1)
        v += __shfl_xor(v, off, 32);
    return v;
}

// ---------------------------------------------------------------------------
// Fold conv weights: u_g = W_cg @ w_pg  (32-vec),  const = b_node + sum_g (b_cg.w_pg + b_pg)
// One wave.
// ---------------------------------------------------------------------------
__global__ void fold_weights_kernel(const float* __restrict__ Wc0, const float* __restrict__ bc0,
                                    const float* __restrict__ Wc1, const float* __restrict__ bc1,
                                    const float* __restrict__ wp0, const float* __restrict__ bp0,
                                    const float* __restrict__ wp1, const float* __restrict__ bp1,
                                    const float* __restrict__ bnode,
                                    float* __restrict__ u0, float* __restrict__ u1,
                                    float* __restrict__ consts) {
    int k = threadIdx.x;  // 0..31
    float s0 = 0.f, s1 = 0.f;
    #pragma unroll
    for (int j = 0; j < H; ++j) {
        s0 += Wc0[k * H + j] * wp0[j];
        s1 += Wc1[k * H + j] * wp1[j];
    }
    u0[k] = s0;
    u1[k] = s1;
    float c0 = wave_reduce_sum(bc0[k] * wp0[k]);
    float c1 = wave_reduce_sum(bc1[k] * wp1[k]);
    if (k == 0) consts[0] = bnode[0] + bp0[0] + bp1[0] + c0 + c1;
}

// ---------------------------------------------------------------------------
// Per-node inverse L2 norm of visual features (one wave32 per node, float4 loads)
// ---------------------------------------------------------------------------
template <int VD>
__global__ void row_inv_norm_kernel(const float* __restrict__ vis, float* __restrict__ invn) {
    int wave = (blockIdx.x * blockDim.x + threadIdx.x) >> 5;
    int lane = threadIdx.x & 31;
    if (wave >= N_NODES) return;
    const float4* v4 = (const float4*)(vis + (size_t)wave * VD);
    float acc = 0.f;
    for (int i = lane; i < VD / 4; i += 32) {
        float4 q = v4[i];
        acc += q.x * q.x + q.y * q.y + q.z * q.z + q.w * q.w;
    }
    acc = wave_reduce_sum(acc);
    if (lane == 0) invn[wave] = rsqrtf(acc + 1e-8f);
}

// ---------------------------------------------------------------------------
// Input MLP fused with WMMA second layer:
//   t = PReLU(BN(x @ W_in1 + b_in1));  h = t @ W_in2 + b_in2
// One wave32 per 16-node tile. K = 32 matches v_wmma_f32_16x16x32_f16 exactly.
//
// A fragment (16-bit A 16x32, ISA 7.12.2): lane L holds row M=L&15,
//   K in {kb..kb+7} U {16+kb..16+kb+7},  kb = (L>>4)*8  -> t computed directly
//   into that K order (no LDS shuffle needed).
// B fragment (16-bit B 32x16): lane L holds col N=L&15, same K order as A:
//   K in {kb2..kb2+7} U {16+kb2..16+kb2+7}, kb2 = (L>>4)*8.
// D (f32 16x16): VGPR r -> row r + (lane>>4)*8, col lane&15.
// ---------------------------------------------------------------------------
__global__ void __launch_bounds__(128) mlp_h_kernel(
        const float* __restrict__ x,
        const float* __restrict__ Win1, const float* __restrict__ bin1,
        const float* __restrict__ gamma, const float* __restrict__ beta,
        const float* __restrict__ mean,  const float* __restrict__ var,
        const float* __restrict__ prelu_a,
        const float* __restrict__ Win2,  const float* __restrict__ bin2,
        float* __restrict__ h) {
    int wave = (blockIdx.x * blockDim.x + threadIdx.x) >> 5;
    int lane = threadIdx.x & 31;
    if (wave >= N_NODES / 16) return;

    const int tile = wave;
    const int row  = lane & 15;
    const int half = lane >> 4;          // 0 or 1 (K-split half)
    const int node = tile * 16 + row;
    const float slope = prelu_a[0];

    float xr[9];
    #pragma unroll
    for (int j = 0; j < 9; ++j) xr[j] = x[node * 9 + j];

    // build A fragment: this lane's 16 K-values of t[row][*]
    const int kbase = half * 8;
    v16h a;
    #pragma unroll
    for (int i = 0; i < 16; ++i) {
        int k = (i < 8) ? (kbase + i) : (16 + kbase + (i - 8));
        float acc = bin1[k];
        #pragma unroll
        for (int j = 0; j < 9; ++j) acc += xr[j] * Win1[j * H + k];
        acc = gamma[k] * (acc - mean[k]) * rsqrtf(var[k] + 1e-5f) + beta[k];
        acc = (acc >= 0.f) ? acc : slope * acc;
        a[i] = (_Float16)acc;
    }

    // B fragments for output columns [0..15] and [16..31]
    const int col = lane & 15;
    const int kb2 = half * 8;
    v16h b0, b1;
    #pragma unroll
    for (int j = 0; j < 16; ++j) {
        int kk = (j < 8) ? (kb2 + j) : (16 + kb2 + (j - 8));
        b0[j] = (_Float16)Win2[kk * H + col];
        b1[j] = (_Float16)Win2[kk * H + col + 16];
    }

    v8f c = {};
    v8f d0 = __builtin_amdgcn_wmma_f32_16x16x32_f16(false, a, false, b0, (short)0, c, false, false);
    v8f d1 = __builtin_amdgcn_wmma_f32_16x16x32_f16(false, a, false, b1, (short)0, c, false, false);

    #pragma unroll
    for (int r = 0; r < 8; ++r) {
        int m = r + half * 8;
        size_t base = (size_t)(tile * 16 + m) * H;
        h[base + col]      = d0[r] + bin2[col];
        h[base + col + 16] = d1[r] + bin2[col + 16];
    }
}

// ---------------------------------------------------------------------------
// Fused edge kernel: one wave32 per edge.
//   w_e = dot(vis[src],vis[dst]) * invn[src] * invn[dst]    (float4 gathers, L2-resident)
//   agg[dst][lane] += w_e * h[src][lane]                    (one f32 atomic per lane)
// ---------------------------------------------------------------------------
template <int VD>
__global__ void __launch_bounds__(256) edge_kernel(
        const int* __restrict__ ei,
        const float* __restrict__ vis,
        const float* __restrict__ invn,
        const float* __restrict__ h,
        float* __restrict__ agg) {
    int wave = (blockIdx.x * blockDim.x + threadIdx.x) >> 5;
    int lane = threadIdx.x & 31;
    if (wave >= N_EDGES) return;

    const int src = ei[wave];
    const int dst = ei[N_EDGES + wave];
    const float4* s4 = (const float4*)(vis + (size_t)src * VD);
    const float4* d4 = (const float4*)(vis + (size_t)dst * VD);

    float acc = 0.f;
    #pragma unroll 4
    for (int i = lane; i < VD / 4; i += 32) {
        float4 qa = s4[i];
        float4 qb = d4[i];
        acc += qa.x * qb.x + qa.y * qb.y + qa.z * qb.z + qa.w * qb.w;
    }
    acc = wave_reduce_sum(acc);
    const float w = acc * invn[src] * invn[dst];
    atomicAdd(&agg[(size_t)dst * H + lane], w * h[(size_t)src * H + lane]);
}

// ---------------------------------------------------------------------------
// Final reduction: out[i] = h[i].w_node + agg0[i].u0 + agg1[i].u1 + const
// One wave32 per node, coalesced loads, butterfly reduce.
// ---------------------------------------------------------------------------
__global__ void __launch_bounds__(256) finalize_kernel(
        const float* __restrict__ h,
        const float* __restrict__ agg0,
        const float* __restrict__ agg1,
        const float* __restrict__ wnode,
        const float* __restrict__ u0,
        const float* __restrict__ u1,
        const float* __restrict__ consts,
        float* __restrict__ out) {
    int wave = (blockIdx.x * blockDim.x + threadIdx.x) >> 5;
    int lane = threadIdx.x & 31;
    if (wave >= N_NODES) return;
    size_t base = (size_t)wave * H;
    float v = h[base + lane] * wnode[lane]
            + agg0[base + lane] * u0[lane]
            + agg1[base + lane] * u1[lane];
    v = wave_reduce_sum(v);
    if (lane == 0) out[wave] = v + consts[0];
}

// ---------------------------------------------------------------------------
extern "C" void kernel_launch(void* const* d_in, const int* in_sizes, int n_in,
                              void* d_out, int out_size, void* d_ws, size_t ws_size,
                              hipStream_t stream) {
    const float* x     = (const float*)d_in[0];
    const float* vis0  = (const float*)d_in[1];
    const float* vis1  = (const float*)d_in[2];
    const int*   ei0   = (const int*)  d_in[3];
    const int*   ei1   = (const int*)  d_in[4];
    const float* Win1  = (const float*)d_in[5];
    const float* bin1  = (const float*)d_in[6];
    const float* gamma = (const float*)d_in[7];
    const float* beta  = (const float*)d_in[8];
    const float* mean  = (const float*)d_in[9];
    const float* var   = (const float*)d_in[10];
    const float* prelu = (const float*)d_in[11];
    const float* Win2  = (const float*)d_in[12];
    const float* bin2  = (const float*)d_in[13];
    const float* wnode = (const float*)d_in[14];
    const float* bnode = (const float*)d_in[15];
    const float* Wc0   = (const float*)d_in[16];
    const float* bc0   = (const float*)d_in[17];
    const float* Wc1   = (const float*)d_in[18];
    const float* bc1   = (const float*)d_in[19];
    const float* wp0   = (const float*)d_in[20];
    const float* bp0   = (const float*)d_in[21];
    const float* wp1   = (const float*)d_in[22];
    const float* bp1   = (const float*)d_in[23];
    float* out = (float*)d_out;

    // workspace layout (floats): h | agg0 | agg1 | invn0 | invn1 | u0 | u1 | consts
    float* ws     = (float*)d_ws;
    float* h      = ws;
    float* agg0   = h     + (size_t)N_NODES * H;
    float* agg1   = agg0  + (size_t)N_NODES * H;
    float* invn0  = agg1  + (size_t)N_NODES * H;
    float* invn1  = invn0 + N_NODES;
    float* u0     = invn1 + N_NODES;
    float* u1     = u0 + H;
    float* consts = u1 + H;

    // zero both agg buffers (contiguous) — graph-capture-safe async memset
    hipMemsetAsync(agg0, 0, 2 * (size_t)N_NODES * H * sizeof(float), stream);

    fold_weights_kernel<<<1, 32, 0, stream>>>(Wc0, bc0, Wc1, bc1, wp0, bp0, wp1, bp1,
                                              bnode, u0, u1, consts);

    {   // per-node inverse norms: one wave per node
        const int tpb = 256, wpb = tpb / 32;
        row_inv_norm_kernel<2048><<<(N_NODES + wpb - 1) / wpb, tpb, 0, stream>>>(vis0, invn0);
        row_inv_norm_kernel<256> <<<(N_NODES + wpb - 1) / wpb, tpb, 0, stream>>>(vis1, invn1);
    }
    {   // h via fused MLP + WMMA: one wave per 16-node tile (625 tiles)
        const int waves = N_NODES / 16, tpb = 128, wpb = tpb / 32;
        mlp_h_kernel<<<(waves + wpb - 1) / wpb, tpb, 0, stream>>>(
            x, Win1, bin1, gamma, beta, mean, var, prelu, Win2, bin2, h);
    }
    {   // fused cosine + scatter: one wave per edge
        const int tpb = 256, wpb = tpb / 32;
        edge_kernel<2048><<<(N_EDGES + wpb - 1) / wpb, tpb, 0, stream>>>(ei0, vis0, invn0, h, agg0);
        edge_kernel<256> <<<(N_EDGES + wpb - 1) / wpb, tpb, 0, stream>>>(ei1, vis1, invn1, h, agg1);
    }
    {   // final per-node reduction
        const int tpb = 256, wpb = tpb / 32;
        finalize_kernel<<<(N_NODES + wpb - 1) / wpb, tpb, 0, stream>>>(
            h, agg0, agg1, wnode, u0, u1, consts, out);
    }
}
